// Attention_59330678227085
// MI455X (gfx1250) — matrix operand order repaired and checked by reading, not compiled, not run on hardware
//
#include <hip/hip_runtime.h>
#include <stdint.h>

// ---------------------------------------------------------------------------
// Types / WMMA helpers (CDNA5 gfx1250, wave32)
// ---------------------------------------------------------------------------
typedef __attribute__((ext_vector_type(16))) _Float16 v16h;
typedef __attribute__((ext_vector_type(8)))  _Float16 v8h;
typedef __attribute__((ext_vector_type(8)))  float    v8f;

#define DEV __device__ __forceinline__

DEV v8f wmma_f16(v16h a, v16h b, v8f c) {
  // D = A(16x32 f16) x B(32x16 f16) + C(16x16 f32)
  return __builtin_amdgcn_wmma_f32_16x16x32_f16(
      /*neg_a=*/false, a, /*neg_b=*/false, b,
      /*c_mod=*/(short)0, c, /*reuse_a=*/false, /*reuse_b=*/false);
}

// A-fragment (16x32, f16): lane l holds row m=l&15;
// halves 0..7  -> k = k0 + (l>=16 ? 16 : 0) + 0..7
// halves 8..15 -> k = k0 + 8 + (l>=16 ? 16 : 0) + 0..7
DEV v16h load_a(const _Float16* base, int ld, int m0, int k0, int lane) {
  const _Float16* row =
      base + (size_t)(m0 + (lane & 15)) * ld + (k0 + ((lane >> 4) << 4));
  v8h lo = *(const v8h*)(row);
  v8h hi = *(const v8h*)(row + 8);
  return __builtin_shufflevector(lo, hi, 0, 1, 2, 3, 4, 5, 6, 7,
                                 8, 9, 10, 11, 12, 13, 14, 15);
}

// B-fragment (32x16, f16) from a TRANSPOSED matrix Bt[n][k]:
// b[h] = B[k][n] with n = l&15, k = k0 + (l>>4)*16 + h  -> one 32B load/lane.
DEV v16h load_bt(const _Float16* base, int ld, int n0, int k0, int lane) {
  const _Float16* p =
      base + (size_t)(n0 + (lane & 15)) * ld + (k0 + ((lane >> 4) << 4));
  return *(const v16h*)p;
}

// CDNA5 async copy: each active lane moves 16B global -> LDS (ASYNCcnt).
DEV void async_ld16B(uint32_t lds_byte, uint64_t gaddr) {
  asm volatile("global_load_async_to_lds_b128 %0, %1, off"
               :: "v"(lds_byte), "v"(gaddr)
               : "memory");
}

DEV uint32_t lds_off(const void* p) {  // LDS aperture: low 32 bits = LDS addr
  return (uint32_t)(uintptr_t)p;
}

// ---------------------------------------------------------------------------
// Problem constants
// ---------------------------------------------------------------------------
#define BATCH 2
#define SEQ   4096
#define CDIM  384
#define HEADS 6
#define HDIM  64
#define ROWS  (BATCH * SEQ)      // 8192
#define QKVN  (3 * CDIM)         // 1152
#define ATT_SCALE 0.125f         // 64^-0.5
#define LDP   72                 // padded LDS row stride (halves): 144B = 9*16B
                                 // -> b128-aligned, 64-bank conflict-free

// ---------------------------------------------------------------------------
// fp32 -> f16 conversion / transpose kernels
// ---------------------------------------------------------------------------
__global__ void cvt_f16(const float* __restrict__ in, _Float16* __restrict__ out,
                        int n) {
  int i = blockIdx.x * blockDim.x + threadIdx.x;
  if (i < n) out[i] = (_Float16)in[i];
}

// in[rows][cols] (fp32) -> out[cols][rows] (f16)
__global__ void cvt_f16_t(const float* __restrict__ in, _Float16* __restrict__ out,
                          int rows, int cols) {
  int i = blockIdx.x * blockDim.x + threadIdx.x;
  if (i < rows * cols) {
    int r = i / cols, c = i - r * cols;
    out[(size_t)c * rows + r] = (_Float16)in[i];
  }
}

// ---------------------------------------------------------------------------
// QKV GEMM: Xh[8192,384] x Wqkv[384,1152] -> scatter into Q,K (B,H,N,D) & Vt (B,H,D,N)
// block = 128 thr = 4 waves (2x2), wave tile = 32(M) x 64(N)
// ---------------------------------------------------------------------------
__global__ __launch_bounds__(128) void qkv_gemm(
    const _Float16* __restrict__ Xh, const _Float16* __restrict__ Wt,
    _Float16* __restrict__ Qh, _Float16* __restrict__ Kh,
    _Float16* __restrict__ Vt) {
  const int lane = threadIdx.x & 31;
  const int wave = threadIdx.x >> 5;
  const int m0 = blockIdx.x * 64 + (wave & 1) * 32;
  const int n0 = blockIdx.y * 128 + (wave >> 1) * 64;

  v8f acc[2][4] = {};
  for (int kk = 0; kk < CDIM; kk += 32) {
    if (kk + 32 < CDIM) {  // gfx1250 global_prefetch_b8
      __builtin_prefetch(Xh + (size_t)(m0 + (lane & 15)) * CDIM + kk + 32);
      __builtin_prefetch(Wt + (size_t)(n0 + (lane & 15)) * CDIM + kk + 32);
    }
    v16h a0 = load_a(Xh, CDIM, m0, kk, lane);
    v16h a1 = load_a(Xh, CDIM, m0 + 16, kk, lane);
#pragma unroll
    for (int nj = 0; nj < 4; ++nj) {
      v16h b = load_bt(Wt, CDIM, n0 + nj * 16, kk, lane);
      acc[0][nj] = wmma_f16(a0, b, acc[0][nj]);
      acc[1][nj] = wmma_f16(a1, b, acc[1][nj]);
    }
  }

  // Epilogue: C/D layout (reg r, lane l) -> (m = r + (l>>4)*8, n = l&15)
#pragma unroll
  for (int mi = 0; mi < 2; ++mi) {
#pragma unroll
    for (int nj = 0; nj < 4; ++nj) {
#pragma unroll
      for (int r = 0; r < 8; ++r) {
        int m = m0 + mi * 16 + r + ((lane >> 4) << 3);  // global row (b*N+tok)
        int j = n0 + nj * 16 + (lane & 15);             // col in [0,1152)
        float v = acc[mi][nj][r];
        int s = j / CDIM;           // 0=Q, 1=K, 2=V
        int rem = j - s * CDIM;
        int h = rem >> 6, d = rem & 63;
        int b = m >> 12, tok = m & (SEQ - 1);
        size_t bh = (size_t)(b * HEADS + h);
        if (s == 0)
          Qh[(bh * SEQ + tok) * HDIM + d] = (_Float16)(v * ATT_SCALE);
        else if (s == 1)
          Kh[(bh * SEQ + tok) * HDIM + d] = (_Float16)v;
        else
          Vt[(bh * HDIM + d) * SEQ + tok] = (_Float16)v;  // transposed store
      }
    }
  }
}

// ---------------------------------------------------------------------------
// Stage one 64-key block of K (64x64, row=key) and V^T (64x64, row=d) into LDS
// with CDNA5 async-to-LDS copies. 256 threads x 2 chunks x 16B each per tile.
// LDS rows padded to LDP halves.
// ---------------------------------------------------------------------------
DEV void stage_kv(const _Float16* __restrict__ Kp, const _Float16* __restrict__ Vp,
                  int kv0, _Float16* kb, _Float16* vb, int tid) {
  const uint32_t kl = lds_off(kb);
  const uint32_t vl = lds_off(vb);
  const _Float16* kg = Kp + (size_t)kv0 * HDIM;  // K rows contiguous (ld=64)
  const _Float16* vg = Vp + kv0;                 // Vt rows stride SEQ
#pragma unroll
  for (int c = 0; c < 2; ++c) {
    int i = tid + c * 256;            // chunk 0..511
    int row = i >> 3, sub = i & 7;    // 64 rows x 8 x 16B
    uint32_t lofs = (uint32_t)(row * (LDP * 2) + sub * 16);
    async_ld16B(kl + lofs, (uint64_t)(uintptr_t)(kg + (size_t)row * HDIM + sub * 8));
    async_ld16B(vl + lofs, (uint64_t)(uintptr_t)(vg + (size_t)row * SEQ + sub * 8));
  }
}

// ---------------------------------------------------------------------------
// Flash attention: grid(32, B*H), 256 thr = 8 waves, wave owns 16 query rows.
// Double-buffered async K/V staging; online softmax over 64-key blocks;
// P relayout C/D->A through per-wave LDS.
// ---------------------------------------------------------------------------
__global__ __launch_bounds__(256) void flash_attn(
    const _Float16* __restrict__ Qh, const _Float16* __restrict__ Kh,
    const _Float16* __restrict__ Vt, _Float16* __restrict__ Ah) {
  __shared__ __align__(16) _Float16 KB[2][64 * LDP];
  __shared__ __align__(16) _Float16 VB[2][64 * LDP];
  __shared__ __align__(16) _Float16 Pl[8][16 * LDP];

  const int tid = threadIdx.x;
  const int lane = tid & 31;
  const int wave = tid >> 5;
  const int bh = blockIdx.y;          // 0..11
  const int b = bh / HEADS;
  const int h = bh - b * HEADS;
  const int row0 = blockIdx.x * 128 + wave * 16;

  const _Float16* Qp = Qh + (size_t)bh * SEQ * HDIM;
  const _Float16* Kp = Kh + (size_t)bh * SEQ * HDIM;
  const _Float16* Vp = Vt + (size_t)bh * HDIM * SEQ;
  _Float16* myP = &Pl[wave][0];

  // Q tile (16x64) in registers: two A-fragments (k=0..31, 32..63)
  v16h qa0 = load_a(Qp, HDIM, row0, 0, lane);
  v16h qa1 = load_a(Qp, HDIM, row0, 32, lane);

  float mrow[8], lrow[8];
  v8f Oacc[4] = {};
#pragma unroll
  for (int r = 0; r < 8; ++r) { mrow[r] = -1e30f; lrow[r] = 0.f; }

  // Prologue: stage first K/V block into buffer 0.
  stage_kv(Kp, Vp, 0, KB[0], VB[0], tid);

  for (int it = 0; it < SEQ / 64; ++it) {
    const int cur = it & 1;
    // Own async copies done; barrier makes all waves' staging visible AND
    // guarantees everyone finished reading the buffer we stage into next.
    asm volatile("s_wait_asynccnt 0x0" ::: "memory");
    __syncthreads();
    if (it + 1 < SEQ / 64)
      stage_kv(Kp, Vp, (it + 1) * 64, KB[cur ^ 1], VB[cur ^ 1], tid);

    const _Float16* kb = KB[cur];
    const _Float16* vb = VB[cur];

    // ---- S = Q * K^T (16 x 64), 4 sub-tiles of 16 cols ----
    v8f S[4];
#pragma unroll
    for (int t = 0; t < 4; ++t) {
      v16h kb0 = load_bt(kb, LDP, t * 16, 0, lane);
      v16h kb1 = load_bt(kb, LDP, t * 16, 32, lane);
      v8f z = {};
      S[t] = wmma_f16(qa0, kb0, z);
      S[t] = wmma_f16(qa1, kb1, S[t]);
    }

    // ---- block row max (reduce over 4 sub-tiles, then 16 lanes) ----
    float bm[8];
#pragma unroll
    for (int r = 0; r < 8; ++r)
      bm[r] = fmaxf(fmaxf(S[0][r], S[1][r]), fmaxf(S[2][r], S[3][r]));
#pragma unroll
    for (int r = 0; r < 8; ++r) {
      bm[r] = fmaxf(bm[r], __shfl_xor(bm[r], 1, 32));
      bm[r] = fmaxf(bm[r], __shfl_xor(bm[r], 2, 32));
      bm[r] = fmaxf(bm[r], __shfl_xor(bm[r], 4, 32));
      bm[r] = fmaxf(bm[r], __shfl_xor(bm[r], 8, 32));
    }

    float mnew[8], alpha[8], bl[8];
#pragma unroll
    for (int r = 0; r < 8; ++r) {
      mnew[r] = fmaxf(mrow[r], bm[r]);
      alpha[r] = __expf(mrow[r] - mnew[r]);
      mrow[r] = mnew[r];
      bl[r] = 0.f;
    }

    // ---- P = exp(S - mnew): accumulate row sums, spill f16 tile to LDS ----
#pragma unroll
    for (int t = 0; t < 4; ++t) {
#pragma unroll
      for (int r = 0; r < 8; ++r) {
        float p = __expf(S[t][r] - mnew[r]);
        bl[r] += p;
        myP[(r + ((lane >> 4) << 3)) * LDP + t * 16 + (lane & 15)] = (_Float16)p;
      }
    }
#pragma unroll
    for (int r = 0; r < 8; ++r) {
      bl[r] += __shfl_xor(bl[r], 1, 32);
      bl[r] += __shfl_xor(bl[r], 2, 32);
      bl[r] += __shfl_xor(bl[r], 4, 32);
      bl[r] += __shfl_xor(bl[r], 8, 32);
      lrow[r] = lrow[r] * alpha[r] + bl[r];
    }

    // ---- rescale O accumulator ----
#pragma unroll
    for (int j = 0; j < 4; ++j)
#pragma unroll
      for (int r = 0; r < 8; ++r) Oacc[j][r] *= alpha[r];

    // Per-wave LDS tile: make the ds stores visible before reloading as A-frags.
    asm volatile("s_wait_dscnt 0x0" ::: "memory");
    v16h pa0 = load_a(myP, LDP, 0, 0, lane);
    v16h pa1 = load_a(myP, LDP, 0, 32, lane);

    // ---- O += P * V  (B-frag from transposed V tile in LDS) ----
#pragma unroll
    for (int j = 0; j < 4; ++j) {
      v16h vb0 = load_bt(vb, LDP, j * 16, 0, lane);
      v16h vb1 = load_bt(vb, LDP, j * 16, 32, lane);
      Oacc[j] = wmma_f16(pa0, vb0, Oacc[j]);
      Oacc[j] = wmma_f16(pa1, vb1, Oacc[j]);
    }
  }

  // ---- normalize and store to Ah[B*N, C] (f16) ----
#pragma unroll
  for (int j = 0; j < 4; ++j) {
#pragma unroll
    for (int r = 0; r < 8; ++r) {
      int m = row0 + r + ((lane >> 4) << 3);
      int n = h * HDIM + j * 16 + (lane & 15);
      Ah[((size_t)b * SEQ + m) * CDIM + n] = (_Float16)(Oacc[j][r] / lrow[r]);
    }
  }
}

// ---------------------------------------------------------------------------
// Output projection: Ah[8192,384] x Wp[384,384] + bias -> out fp32
// ---------------------------------------------------------------------------
__global__ __launch_bounds__(128) void proj_gemm(
    const _Float16* __restrict__ Ah, const _Float16* __restrict__ Wpt,
    const float* __restrict__ bias, float* __restrict__ out) {
  const int lane = threadIdx.x & 31;
  const int wave = threadIdx.x >> 5;
  const int m0 = blockIdx.x * 64 + (wave & 1) * 32;
  const int n0 = blockIdx.y * 128 + (wave >> 1) * 64;

  v8f acc[2][4] = {};
  for (int kk = 0; kk < CDIM; kk += 32) {
    if (kk + 32 < CDIM) {
      __builtin_prefetch(Ah + (size_t)(m0 + (lane & 15)) * CDIM + kk + 32);
      __builtin_prefetch(Wpt + (size_t)(n0 + (lane & 15)) * CDIM + kk + 32);
    }
    v16h a0 = load_a(Ah, CDIM, m0, kk, lane);
    v16h a1 = load_a(Ah, CDIM, m0 + 16, kk, lane);
#pragma unroll
    for (int nj = 0; nj < 4; ++nj) {
      v16h bfr = load_bt(Wpt, CDIM, n0 + nj * 16, kk, lane);
      acc[0][nj] = wmma_f16(a0, bfr, acc[0][nj]);
      acc[1][nj] = wmma_f16(a1, bfr, acc[1][nj]);
    }
  }

#pragma unroll
  for (int mi = 0; mi < 2; ++mi) {
#pragma unroll
    for (int nj = 0; nj < 4; ++nj) {
#pragma unroll
      for (int r = 0; r < 8; ++r) {
        int m = m0 + mi * 16 + r + ((lane >> 4) << 3);
        int n = n0 + nj * 16 + (lane & 15);
        out[(size_t)m * CDIM + n] = acc[mi][nj][r] + bias[n];
      }
    }
  }
}

// ---------------------------------------------------------------------------
// Host launcher
// ---------------------------------------------------------------------------
extern "C" void kernel_launch(void* const* d_in, const int* in_sizes, int n_in,
                              void* d_out, int out_size, void* d_ws,
                              size_t ws_size, hipStream_t stream) {
  const float* x      = (const float*)d_in[0];   // [B,N,C]
  const float* w_qkv  = (const float*)d_in[1];   // [C,3C]
  const float* w_proj = (const float*)d_in[2];   // [C,C]
  const float* b_proj = (const float*)d_in[3];   // [C]
  float* out = (float*)d_out;

  // Workspace layout (bytes, all 256-aligned)
  char* ws = (char*)d_ws;
  const size_t SZ_X   = (size_t)ROWS * CDIM * 2;          // 6,291,456
  const size_t SZ_WQ  = (size_t)QKVN * CDIM * 2;          //   884,736
  const size_t SZ_WP  = (size_t)CDIM * CDIM * 2;          //   294,912
  const size_t SZ_QKV = (size_t)BATCH * HEADS * SEQ * HDIM * 2;  // 6,291,456

  _Float16* Xh    = (_Float16*)(ws);
  _Float16* Wqkvt = (_Float16*)(ws + SZ_X);
  _Float16* Wpt   = (_Float16*)(ws + SZ_X + SZ_WQ);
  _Float16* Qh    = (_Float16*)(ws + SZ_X + SZ_WQ + SZ_WP);
  _Float16* Kh    = (_Float16*)(ws + SZ_X + SZ_WQ + SZ_WP + SZ_QKV);
  _Float16* Vt    = (_Float16*)(ws + SZ_X + SZ_WQ + SZ_WP + 2 * SZ_QKV);
  _Float16* Ah    = (_Float16*)(ws + SZ_X + SZ_WQ + SZ_WP + 3 * SZ_QKV);

  // 1) fp32 -> f16 (weights transposed for contiguous B-fragment loads)
  cvt_f16<<<(ROWS * CDIM + 255) / 256, 256, 0, stream>>>(x, Xh, ROWS * CDIM);
  cvt_f16_t<<<(CDIM * QKVN + 255) / 256, 256, 0, stream>>>(w_qkv, Wqkvt, CDIM, QKVN);
  cvt_f16_t<<<(CDIM * CDIM + 255) / 256, 256, 0, stream>>>(w_proj, Wpt, CDIM, CDIM);

  // 2) QKV projection -> Q (pre-scaled), K, V^T
  qkv_gemm<<<dim3(ROWS / 64, QKVN / 128), 128, 0, stream>>>(Xh, Wqkvt, Qh, Kh, Vt);

  // 3) flash attention (async double-buffered K/V staging) -> Ah[B*N, C] f16
  flash_attn<<<dim3(SEQ / 128, BATCH * HEADS), 256, 0, stream>>>(Qh, Kh, Vt, Ah);

  // 4) output projection + bias -> fp32 out
  proj_gemm<<<dim3(ROWS / 64, CDIM / 128), 128, 0, stream>>>(Ah, Wpt, b_proj, out);
}
